// LSTMClassifier_89730456748206
// MI455X (gfx1250) — matrix operand order repaired and checked by reading, not compiled, not run on hardware
//
#include <hip/hip_runtime.h>

// ---------------------------------------------------------------------------
// 2-layer LSTM (B=1024, T=512, I=128, H=64) + FC(64->1) + sigmoid, MI455X.
//
// Roofline: ~86 GFLOP, but only ~256 MB mandatory HBM traffic (x read once)
// => ~11us at 23.3 TB/s; the real limit is the 512-step serial recurrence.
// Design: one fused persistent kernel, 64 workgroups x 16 batch rows, both
// LSTM layers advanced per step. All weight (B) WMMA fragments are hoisted
// into 160 persistent VGPRs per wave before the time loop, so the per-step
// critical path is only: barrier -> A-fragment ds_load_b128 -> 20x
// v_wmma_f32_16x16x32_bf16 (register-fed B) -> gate LDS stores -> cell math.
// x is software-prefetched one step ahead into registers.
// ---------------------------------------------------------------------------

#define BATCH 1024
#define SEQ   512
#define IDIM  128
#define HID   64
#define GATES 256            // 4*HID
#define K0TOT 192            // IDIM + HID (layer 0 combined K)
#define K1TOT 128            // HID + HID  (layer 1 combined K)
#define NCH0  6              // K0TOT / 32
#define NCH1  4              // K1TOT / 32
#define BM    16             // batch rows per workgroup

typedef __attribute__((ext_vector_type(16))) __bf16 v16bf;
typedef __attribute__((ext_vector_type(8)))  __bf16 v8bf;
typedef __attribute__((ext_vector_type(8)))  float  v8f;

union AF { v16bf v; v8bf h[2]; };

__device__ __forceinline__ unsigned short f2bf(float f) {
  unsigned int u = __float_as_uint(f);
  u += 0x7FFFu + ((u >> 16) & 1u);        // round-to-nearest-even
  return (unsigned short)(u >> 16);
}
__device__ __forceinline__ float sigf(float x) {
  return 1.0f / (1.0f + __expf(-x));
}
__device__ __forceinline__ float tanhfast(float x) {
  x = fminf(15.0f, fmaxf(-15.0f, x));
  float e = __expf(2.0f * x);
  return (e - 1.0f) / (e + 1.0f);
}

// ---------------------------------------------------------------------------
// Weight packing: W0p[g][k] bf16 with k = [W_ih0 (128) | W_hh0 (64)],
//                 W1p[g][k] bf16 with k = [W_ih1 (64)  | W_hh1 (64)],
//                 biasL[n] = b_ih[n] + b_hh[n].
// Row-major [GATES][K] feeds this code's contiguous-K B fragments directly: lane n reads
// 16 consecutive K values of row n (lane>>4 selects the K-half).
// ---------------------------------------------------------------------------
__global__ void pack_weights(const float* __restrict__ Wih0, const float* __restrict__ Whh0,
                             const float* __restrict__ bih0, const float* __restrict__ bhh0,
                             const float* __restrict__ Wih1, const float* __restrict__ Whh1,
                             const float* __restrict__ bih1, const float* __restrict__ bhh1,
                             unsigned short* __restrict__ W0p, unsigned short* __restrict__ W1p,
                             float* __restrict__ bias0, float* __restrict__ bias1) {
  const int n0 = GATES * K0TOT;
  const int n1 = GATES * K1TOT;
  int idx = blockIdx.x * blockDim.x + threadIdx.x;
  if (idx < n0) {
    int g = idx / K0TOT, k = idx % K0TOT;
    float v = (k < IDIM) ? Wih0[g * IDIM + k] : Whh0[g * HID + (k - IDIM)];
    W0p[idx] = f2bf(v);
  } else if (idx < n0 + n1) {
    int j = idx - n0;
    int g = j / K1TOT, k = j % K1TOT;
    float v = (k < HID) ? Wih1[g * HID + k] : Whh1[g * HID + (k - HID)];
    W1p[j] = f2bf(v);
  } else if (idx < n0 + n1 + GATES) {
    int n = idx - n0 - n1;
    bias0[n] = bih0[n] + bhh0[n];
  } else if (idx < n0 + n1 + 2 * GATES) {
    int n = idx - n0 - n1 - GATES;
    bias1[n] = bih1[n] + bhh1[n];
  }
}

// GEMM slab: one wave computes 16(M) x 32(N) of the gate matrix with
// register-resident B fragments. A-fragment matches B: lane half picks
// K 0-15 vs 16-31 -> two contiguous 16B LDS loads.
template <int NCH>
__device__ __forceinline__ void gemm_tiles_reg(const __bf16* __restrict__ sA,
                                               const AF (&w0)[NCH], const AF (&w1)[NCH],
                                               float bv0, float bv1,
                                               float* __restrict__ sG,
                                               int lane, int wave) {
  constexpr int KTOT = 32 * NCH;
  v8f acc0 = {};
  v8f acc1 = {};
  const int mrow = lane & 15;
  const int hi   = lane >> 4;
  const __bf16* arow = sA + mrow * KTOT + 16 * hi;
#pragma unroll
  for (int kc = 0; kc < NCH; ++kc) {
    AF a;
    a.h[0] = *(const v8bf*)(arow + kc * 32);
    a.h[1] = *(const v8bf*)(arow + kc * 32 + 8);
    acc0 = __builtin_amdgcn_wmma_f32_16x16x32_bf16(false, a.v, false, w0[kc].v,
                                                   (short)0, acc0, false, false);
    acc1 = __builtin_amdgcn_wmma_f32_16x16x32_bf16(false, a.v, false, w1[kc].v,
                                                   (short)0, acc1, false, false);
  }
  // D layout: VGPR j holds row M = j + 8*(lane>=16), col N = lane&15.
  const int nc0 = wave * 32 + (lane & 15);
  const int nc1 = nc0 + 16;
#pragma unroll
  for (int j = 0; j < 8; ++j) {
    const int row = j + 8 * hi;
    sG[row * GATES + nc0] = acc0[j] + bv0;
    sG[row * GATES + nc1] = acc1[j] + bv1;
  }
}

// Gate nonlinearity + state update: 16x64 items, 4 per thread.
__device__ __forceinline__ void lstm_cell(const float* __restrict__ sG,
                                          float* __restrict__ sC,
                                          unsigned short* __restrict__ hdst0, int stride0,
                                          unsigned short* __restrict__ hdst1, int stride1,
                                          float* __restrict__ sH, int tid) {
  const int m  = tid >> 4;
  const int jb = (tid & 15) * 4;
#pragma unroll
  for (int q = 0; q < 4; ++q) {
    const int j = jb + q;
    const float gi = sG[m * GATES + j];
    const float gf = sG[m * GATES + 64 + j];
    const float gc = sG[m * GATES + 128 + j];
    const float go = sG[m * GATES + 192 + j];
    const float iv = sigf(gi);
    const float fv = sigf(gf);
    const float cv = tanhfast(gc);
    const float ov = sigf(go);
    const float c  = fv * sC[m * HID + j] + iv * cv;
    sC[m * HID + j] = c;
    const float h = ov * tanhfast(c);
    const unsigned short hb = f2bf(h);
    hdst0[m * stride0 + j] = hb;
    if (hdst1) hdst1[m * stride1 + j] = hb;
    if (sH) sH[m * HID + j] = h;
  }
}

__global__ __launch_bounds__(256) void lstm_fused(
    const float* __restrict__ x,               // [B][T][128]
    const unsigned short* __restrict__ W0p,    // [256][192] bf16 bits
    const unsigned short* __restrict__ W1p,    // [256][128] bf16 bits
    const float* __restrict__ bias0,           // [256]
    const float* __restrict__ bias1,           // [256]
    const float* __restrict__ fcw,             // [64]
    const float* __restrict__ fcb,             // [1]
    float* __restrict__ out) {                 // [1024]
  __shared__ __align__(16) __bf16 sA0[BM * K0TOT];   // [x_t | h0_{t-1}]
  __shared__ __align__(16) __bf16 sA1[BM * K1TOT];   // [h0_t | h1_{t-1}]
  __shared__ float sG[BM * GATES];                   // gate buffer (both layers)
  __shared__ float sC0[BM * HID];
  __shared__ float sC1[BM * HID];
  __shared__ float sH1[BM * HID];                    // f32 h1 for final FC

  const int tid  = threadIdx.x;
  const int lane = tid & 31;
  const int wave = tid >> 5;
  const int b0   = blockIdx.x * BM;

  // --- hoist all weight B-fragments + biases into registers ---------------
  // B-fragment: lane&15 = column (gate row), lane>>4 = K-half; per lane one
  // contiguous 32B read from the packed row-major weight matrix.
  AF w00[NCH0], w01[NCH0];   // layer 0, N-tiles 0/1 of this wave
  AF w10[NCH1], w11[NCH1];   // layer 1
  {
    const int hi  = lane >> 4;
    const int nc0 = wave * 32 + (lane & 15);
    const __bf16* p00 = (const __bf16*)W0p + nc0 * K0TOT + 16 * hi;
    const __bf16* p01 = p00 + 16 * K0TOT;
#pragma unroll
    for (int kc = 0; kc < NCH0; ++kc) {
      w00[kc].h[0] = *(const v8bf*)(p00 + kc * 32);
      w00[kc].h[1] = *(const v8bf*)(p00 + kc * 32 + 8);
      w01[kc].h[0] = *(const v8bf*)(p01 + kc * 32);
      w01[kc].h[1] = *(const v8bf*)(p01 + kc * 32 + 8);
    }
    const __bf16* p10 = (const __bf16*)W1p + nc0 * K1TOT + 16 * hi;
    const __bf16* p11 = p10 + 16 * K1TOT;
#pragma unroll
    for (int kc = 0; kc < NCH1; ++kc) {
      w10[kc].h[0] = *(const v8bf*)(p10 + kc * 32);
      w10[kc].h[1] = *(const v8bf*)(p10 + kc * 32 + 8);
      w11[kc].h[0] = *(const v8bf*)(p11 + kc * 32);
      w11[kc].h[1] = *(const v8bf*)(p11 + kc * 32 + 8);
    }
  }
  const int ncb  = wave * 32 + (lane & 15);
  const float bv00 = bias0[ncb];
  const float bv01 = bias0[ncb + 16];
  const float bv10 = bias1[ncb];
  const float bv11 = bias1[ncb + 16];

  // --- zero initial h/c state ----------------------------------------------
  for (int i = tid; i < BM * HID; i += 256) {
    const int m = i / HID, j = i % HID;
    sC0[i] = 0.0f;
    sC1[i] = 0.0f;
    ((unsigned short*)sA0)[m * K0TOT + IDIM + j] = 0;      // h0_{-1}
    ((unsigned short*)sA1)[m * K1TOT + j]        = 0;      // h0 slot
    ((unsigned short*)sA1)[m * K1TOT + HID + j]  = 0;      // h1_{-1}
  }

  // --- x prefetch registers: 8 contiguous floats per thread ----------------
  const int m_x  = tid >> 4;           // 0..15
  const int i0_x = (tid & 15) * 8;     // 0..120
  const float* xrow = x + ((size_t)(b0 + m_x) * SEQ) * IDIM + i0_x;
  float4 xa = ((const float4*)xrow)[0];
  float4 xb = ((const float4*)xrow)[1];

  for (int t = 0; t < SEQ; ++t) {
    // stage x_t (bf16) into layer-0 A buffer
    {
      unsigned short* dst = (unsigned short*)sA0 + m_x * K0TOT + i0_x;
      dst[0] = f2bf(xa.x); dst[1] = f2bf(xa.y);
      dst[2] = f2bf(xa.z); dst[3] = f2bf(xa.w);
      dst[4] = f2bf(xb.x); dst[5] = f2bf(xb.y);
      dst[6] = f2bf(xb.z); dst[7] = f2bf(xb.w);
    }
    __syncthreads();

    // prefetch x_{t+1}: loads stay in flight under the WMMA work below
    if (t + 1 < SEQ) {
      const float* p = xrow + (size_t)(t + 1) * IDIM;
      xa = ((const float4*)p)[0];
      xb = ((const float4*)p)[1];
    }

    // ---- layer 0: gates = [x_t | h0_{t-1}] @ W0^T + bias (K = 192) -------
    gemm_tiles_reg<NCH0>(sA0, w00, w01, bv00, bv01, sG, lane, wave);
    __syncthreads();
    lstm_cell(sG, sC0,
              (unsigned short*)sA0 + IDIM, K0TOT,   // h0_t for next step
              (unsigned short*)sA1, K1TOT,          // h0_t as layer-1 input
              nullptr, tid);
    __syncthreads();

    // ---- layer 1: gates = [h0_t | h1_{t-1}] @ W1^T + bias (K = 128) ------
    gemm_tiles_reg<NCH1>(sA1, w10, w11, bv10, bv11, sG, lane, wave);
    __syncthreads();
    lstm_cell(sG, sC1,
              (unsigned short*)sA1 + HID, K1TOT,    // h1_t for next step
              nullptr, 0,
              sH1, tid);                            // f32 copy for final FC
    __syncthreads();
  }

  // ---- final FC + sigmoid on h1_{T-1} -------------------------------------
  if (tid < BM) {
    float acc = fcb[0];
#pragma unroll 8
    for (int j = 0; j < HID; ++j) acc += sH1[tid * HID + j] * fcw[j];
    out[b0 + tid] = sigf(acc);
  }
}

extern "C" void kernel_launch(void* const* d_in, const int* in_sizes, int n_in,
                              void* d_out, int out_size, void* d_ws, size_t ws_size,
                              hipStream_t stream) {
  const float* x    = (const float*)d_in[0];
  const float* Wih0 = (const float*)d_in[1];
  const float* Whh0 = (const float*)d_in[2];
  const float* bih0 = (const float*)d_in[3];
  const float* bhh0 = (const float*)d_in[4];
  const float* Wih1 = (const float*)d_in[5];
  const float* Whh1 = (const float*)d_in[6];
  const float* bih1 = (const float*)d_in[7];
  const float* bhh1 = (const float*)d_in[8];
  const float* fcw  = (const float*)d_in[9];
  const float* fcb  = (const float*)d_in[10];

  // workspace: packed bf16 weights + fused biases (~166 KB)
  unsigned short* W0p = (unsigned short*)d_ws;              // 256*192
  unsigned short* W1p = W0p + GATES * K0TOT;                // 256*128
  float* bias0 = (float*)(W1p + GATES * K1TOT);             // 256
  float* bias1 = bias0 + GATES;                             // 256

  const int total = GATES * K0TOT + GATES * K1TOT + 2 * GATES;
  pack_weights<<<(total + 255) / 256, 256, 0, stream>>>(
      Wih0, Whh0, bih0, bhh0, Wih1, Whh1, bih1, bhh1, W0p, W1p, bias0, bias1);

  lstm_fused<<<BATCH / BM, 256, 0, stream>>>(
      x, W0p, W1p, bias0, bias1, fcw, fcb, (float*)d_out);
}
